// MixtureFFN_13408887898973
// MI455X (gfx1250) — matrix operand order repaired and checked by reading, not compiled, not run on hardware
//
#include <hip/hip_runtime.h>
#include <math.h>

#define M_EXP 16
#define INP   64
#define HID   128
#define OUTD  32
#define COEFW 0.25f

// d_ws layout (units: ushort/bf16 elements), all transposed [N][K] bf16:
#define WS_MW1  0                      // [128][64]
#define WS_MW2  8192                   // [16][128]
#define WS_EW1  10240                  // 16 x [128][64]
#define WS_EW2  141312                 // 16 x [32][128]
#define WS_TOT  206848                 // elements
#define WS_BYTES ((size_t)WS_TOT * 2)

typedef __attribute__((ext_vector_type(16))) __bf16 v16bf;
typedef __attribute__((ext_vector_type(8)))  float  v8f;

union BF16x16 { v16bf v; unsigned int u[8]; };

__device__ __forceinline__ unsigned short f2bf(float f) {
  union { __bf16 h; unsigned short s; } c;
  c.h = (__bf16)f;                       // native v_cvt_*_bf16_f32
  return c.s;
}
__device__ __forceinline__ unsigned int pk2(float a, float b) {
  union { unsigned int u; __bf16 h[2]; } c;
  c.h[0] = (__bf16)a; c.h[1] = (__bf16)b;  // should fuse to v_cvt_pk_bf16_f32
  return c.u;
}

__device__ __forceinline__ float redsum16(float v) {
  v += __shfl_xor(v, 1); v += __shfl_xor(v, 2);
  v += __shfl_xor(v, 4); v += __shfl_xor(v, 8);
  return v;
}
__device__ __forceinline__ float redmax16(float v) {
  v = fmaxf(v, __shfl_xor(v, 1)); v = fmaxf(v, __shfl_xor(v, 2));
  v = fmaxf(v, __shfl_xor(v, 4)); v = fmaxf(v, __shfl_xor(v, 8));
  return v;
}

__device__ __forceinline__ v8f wmma_bf16(v16bf a, v16bf b, v8f c) {
  return __builtin_amdgcn_wmma_f32_16x16x32_bf16(false, a, false, b,
                                                 (short)0, c, false, false);
}

// A-fragment (16x32 bf16) of x tile straight from global f32 (row stride INP)
__device__ __forceinline__ v16bf loadA_x(const float* x, int row0, int lane, int kc) {
  BF16x16 r;
  const float* xr = x + (size_t)(row0 + (lane & 15)) * INP;
  int kb = kc * 32 + ((lane >> 4) * 8);
  #pragma unroll
  for (int v = 0; v < 8; ++v) {
    int k = kb + (v & 3) * 2 + (v >> 2) * 16;
    float2 f = *(const float2*)(xr + k);
    r.u[v] = pk2(f.x, f.y);
  }
  return r.v;
}

// A-fragment from LDS bf16 row-major [16][HID]
__device__ __forceinline__ v16bf loadA_h(const unsigned int* h32, int lane, int kc) {
  BF16x16 r;
  int row = lane & 15;
  int kb = kc * 32 + ((lane >> 4) * 8);
  #pragma unroll
  for (int v = 0; v < 8; ++v) {
    int k = kb + (v & 3) * 2 + (v >> 2) * 16;
    r.u[v] = h32[row * (HID / 2) + (k >> 1)];
  }
  return r.v;
}

// B-fragment (32x16 bf16) from LDS weights stored transposed [N][K]
__device__ __forceinline__ v16bf loadB(const unsigned int* w32, int Kstride,
                                       int lane, int nt, int kc) {
  BF16x16 r;
  int n = nt * 16 + (lane & 15);
  int base = (n * Kstride + kc * 32 + ((lane >> 4) * 8)) >> 1;
  #pragma unroll
  for (int v = 0; v < 8; ++v) r.u[v] = w32[base + (v & 3) + (v >> 2) * 8];
  return r.v;
}

// ---- one-shot: convert all weights to bf16, transposed [N][K], into d_ws ----
__global__ __launch_bounds__(256)
void convw_kernel(const float* __restrict__ mw1, const float* __restrict__ mw2,
                  const float* __restrict__ ew1, const float* __restrict__ ew2,
                  __bf16* __restrict__ ws) {
  for (int o = blockIdx.x * 256 + threadIdx.x; o < WS_TOT; o += gridDim.x * 256) {
    float v;
    if (o < WS_MW2) {                       // mw1 [64][128] -> [n=128][k=64]
      int n = o >> 6, k = o & 63;
      v = mw1[k * HID + n];
    } else if (o < WS_EW1) {                // mw2 [128][16] -> [n=16][k=128]
      int i = o - WS_MW2; int n = i >> 7, k = i & 127;
      v = mw2[k * M_EXP + n];
    } else if (o < WS_EW2) {                // ew1 [m][64][128] -> [m][n=128][k=64]
      int i = o - WS_EW1; int m = i >> 13, r = i & 8191;
      int n = r >> 6, k = r & 63;
      v = ew1[m * (INP * HID) + k * HID + n];
    } else {                                // ew2 [m][128][32] -> [m][n=32][k=128]
      int i = o - WS_EW2; int m = i >> 12, r = i & 4095;
      int n = r >> 7, k = r & 127;
      v = ew2[m * (HID * OUTD) + k * OUTD + n];
    }
    ws[o] = (__bf16)v;
  }
}

template<bool PRE>
__global__ __launch_bounds__(256)
void mixffn_kernel(const float* __restrict__ x,   const float* __restrict__ y,
                   const float* __restrict__ mw1, const float* __restrict__ mb1,
                   const float* __restrict__ mw2, const float* __restrict__ mb2,
                   const float* __restrict__ ew1, const float* __restrict__ eb1,
                   const float* __restrict__ ew2, const float* __restrict__ eb2,
                   const unsigned short* __restrict__ wsbf,
                   float* __restrict__ out, int Bsz)
{
  __shared__ __align__(16) unsigned short w1T[HID * INP];   // [n=128][k=64] bf16
  __shared__ __align__(16) unsigned short w2T[OUTD * HID];  // [n<=32][k=128] bf16
  __shared__ __align__(16) float b1s[HID];
  __shared__ __align__(16) float b2s[OUTD];
  __shared__ __align__(16) unsigned short hT[8][16 * HID];  // per-wave h tile bf16

  const int tid  = threadIdx.x;
  const int wave = tid >> 5;
  const int lane = tid & 31;
  const int row0 = blockIdx.x * 128 + wave * 16;
  const int half = lane >> 4;                   // 0: rows 0-7, 1: rows 8-15
  const int nl   = lane & 15;
  if (row0 >= Bsz) return;

  const size_t o_predB = 1;
  const size_t o_idxB  = o_predB + (size_t)Bsz * OUTD;
  const size_t o_predS = o_idxB  + (size_t)Bsz;
  const size_t o_idxS  = o_predS + (size_t)Bsz * OUTD;
  const size_t o_hyp   = o_idxS  + (size_t)Bsz;
  const size_t o_coef  = o_hyp   + (size_t)Bsz * M_EXP * OUTD;

  // ---- stage gating weights ----
  if (PRE) {
    const uint4* s1 = (const uint4*)(wsbf + WS_MW1);
    uint4*       d1 = (uint4*)w1T;
    for (int i = tid; i < (INP * HID) / 8; i += 256) d1[i] = s1[i];
    const uint4* s2 = (const uint4*)(wsbf + WS_MW2);
    uint4*       d2 = (uint4*)w2T;
    for (int i = tid; i < (HID * M_EXP) / 8; i += 256) d2[i] = s2[i];
  } else {
    for (int i = tid; i < INP * HID; i += 256) {
      int k = i / HID, n = i % HID;
      w1T[n * INP + k] = f2bf(mw1[i]);
    }
    for (int i = tid; i < HID * M_EXP; i += 256) {
      int k = i / M_EXP, n = i % M_EXP;
      w2T[n * HID + k] = f2bf(mw2[i]);
    }
  }
  for (int i = tid; i < HID;   i += 256) b1s[i] = mb1[i];
  for (int i = tid; i < M_EXP; i += 256) b2s[i] = mb2[i];
  __syncthreads();

  const unsigned int* w1_32 = (const unsigned int*)w1T;
  const unsigned int* w2_32 = (const unsigned int*)w2T;
  unsigned short* hw   = &hT[wave][0];
  unsigned int*   hw32 = (unsigned int*)hw;

  // x A-fragments: reused by the gating MLP and all 16 experts
  v16bf xA0 = loadA_x(x, row0, lane, 0);
  v16bf xA1 = loadA_x(x, row0, lane, 1);

  // y tile (reused for all 16 dist computations)
  float yv[2][8];
  #pragma unroll
  for (int t = 0; t < 2; ++t)
    #pragma unroll
    for (int j = 0; j < 8; ++j)
      yv[t][j] = y[(size_t)(row0 + half * 8 + j) * OUTD + t * 16 + nl];

  // ---- gating layer 1: [16,64] x [64,128] ----
  v8f acc[8] = {};
  #pragma unroll
  for (int t = 0; t < 8; ++t) {
    v16bf b0 = loadB(w1_32, INP, lane, t, 0);
    acc[t] = wmma_bf16(xA0, b0, acc[t]);
    v16bf b1 = loadB(w1_32, INP, lane, t, 1);
    acc[t] = wmma_bf16(xA1, b1, acc[t]);
  }
  #pragma unroll
  for (int t = 0; t < 8; ++t) {
    int col = t * 16 + nl;
    float bias = b1s[col];
    #pragma unroll
    for (int j = 0; j < 8; ++j)
      hw[(half * 8 + j) * HID + col] = f2bf(fmaxf(acc[t][j] + bias, 0.0f));
  }
  asm volatile("s_wait_dscnt 0x0" ::: "memory");

  // ---- gating layer 2: [16,128] x [128,16] ----
  v8f lg = {};
  #pragma unroll
  for (int kc = 0; kc < 4; ++kc) {
    v16bf a = loadA_h(hw32, lane, kc);
    v16bf b = loadB(w2_32, HID, lane, 0, kc);
    lg = wmma_bf16(a, b, lg);
  }
  {
    float bias = b2s[nl];
    #pragma unroll
    for (int j = 0; j < 8; ++j) lg[j] += bias;
  }

  // per-row softmax stats + greedy argmax; emit coef_logits
  float lse[8]; int sIdx[8];
  #pragma unroll
  for (int j = 0; j < 8; ++j) {
    int row = row0 + half * 8 + j;
    out[o_coef + (size_t)row * M_EXP + nl] = lg[j];
    float mx = redmax16(lg[j]);
    float se = redsum16(expf(lg[j] - mx));
    lse[j] = mx + logf(se);
    float av = lg[j]; int ai = nl;
    #pragma unroll
    for (int msk = 1; msk < 16; msk <<= 1) {
      float ov = __shfl_xor(av, msk);
      int   oi = __shfl_xor(ai, msk);
      if (ov > av || (ov == av && oi < ai)) { av = ov; ai = oi; }
    }
    sIdx[j] = ai;
  }

  // ---- expert loop ----
  float bestD[8]; int bestI[8];
  float predB[2][8], predS[2][8];
  #pragma unroll
  for (int j = 0; j < 8; ++j) {
    bestD[j] = 3.0e38f; bestI[j] = 0;
    predB[0][j] = predB[1][j] = predS[0][j] = predS[1][j] = 0.0f;
  }

  for (int m = 0; m < M_EXP; ++m) {
    __syncthreads();
    if (PRE) {
      const uint4* s1 = (const uint4*)(wsbf + WS_EW1 + m * (INP * HID));
      uint4*       d1 = (uint4*)w1T;
      for (int i = tid; i < (INP * HID) / 8; i += 256) d1[i] = s1[i];
      const uint4* s2 = (const uint4*)(wsbf + WS_EW2 + m * (HID * OUTD));
      uint4*       d2 = (uint4*)w2T;
      for (int i = tid; i < (HID * OUTD) / 8; i += 256) d2[i] = s2[i];
      if (m + 1 < M_EXP)
        __builtin_prefetch(wsbf + WS_EW1 + (m + 1) * (INP * HID) + tid * 32, 0, 1);
    } else {
      const float* W1 = ew1 + (size_t)m * INP * HID;
      const float* W2 = ew2 + (size_t)m * HID * OUTD;
      for (int i = tid; i < INP * HID; i += 256) {
        int k = i / HID, n = i % HID;
        w1T[n * INP + k] = f2bf(W1[i]);
      }
      for (int i = tid; i < HID * OUTD; i += 256) {
        int k = i / OUTD, n = i % OUTD;
        w2T[n * HID + k] = f2bf(W2[i]);
      }
      if (m + 1 < M_EXP)
        __builtin_prefetch(ew1 + (size_t)(m + 1) * INP * HID + tid * 32, 0, 1);
    }
    for (int i = tid; i < HID;  i += 256) b1s[i] = eb1[m * HID + i];
    for (int i = tid; i < OUTD; i += 256) b2s[i] = eb2[m * OUTD + i];
    __syncthreads();

    // layer 1: h = relu(x @ W1 + b1)
    v8f hacc[8] = {};
    #pragma unroll
    for (int t = 0; t < 8; ++t) {
      v16bf b0 = loadB(w1_32, INP, lane, t, 0);
      hacc[t] = wmma_bf16(xA0, b0, hacc[t]);
      v16bf b1 = loadB(w1_32, INP, lane, t, 1);
      hacc[t] = wmma_bf16(xA1, b1, hacc[t]);
    }
    #pragma unroll
    for (int t = 0; t < 8; ++t) {
      int col = t * 16 + nl;
      float bias = b1s[col];
      #pragma unroll
      for (int j = 0; j < 8; ++j)
        hw[(half * 8 + j) * HID + col] = f2bf(fmaxf(hacc[t][j] + bias, 0.0f));
    }
    asm volatile("s_wait_dscnt 0x0" ::: "memory");

    // layer 2: hyp = h @ W2 + b2
    v8f hv[2] = {};
    #pragma unroll
    for (int kc = 0; kc < 4; ++kc) {
      v16bf a = loadA_h(hw32, lane, kc);
      #pragma unroll
      for (int t = 0; t < 2; ++t) {
        v16bf b = loadB(w2_32, HID, lane, t, kc);
        hv[t] = wmma_bf16(a, b, hv[t]);
      }
    }

    float part[8];
    #pragma unroll
    for (int j = 0; j < 8; ++j) part[j] = 0.0f;
    #pragma unroll
    for (int t = 0; t < 2; ++t) {
      int col = t * 16 + nl;
      float bias = b2s[col];
      #pragma unroll
      for (int j = 0; j < 8; ++j) {
        float v = hv[t][j] + bias;
        hv[t][j] = v;
        int row = row0 + half * 8 + j;
        out[o_hyp + ((size_t)row * M_EXP + m) * OUTD + col] = v;
        float d = v - yv[t][j];
        part[j] += d * d;
      }
    }
    #pragma unroll
    for (int j = 0; j < 8; ++j) {
      float dist = redsum16(part[j]);            // sum over all 32 output cols
      bool better = dist < bestD[j];
      if (better) { bestD[j] = dist; bestI[j] = m; }
      bool samp = (m == sIdx[j]);
      #pragma unroll
      for (int t = 0; t < 2; ++t) {
        if (better) predB[t][j] = hv[t][j];
        if (samp)   predS[t][j] = hv[t][j];
      }
    }
  }

  // ---- epilogue: gather best logit, emit preds/indices/loss ----
  float lb[8];
  #pragma unroll
  for (int j = 0; j < 8; ++j) {
    float sel = (nl == bestI[j]) ? lg[j] : 0.0f;
    lb[j] = redsum16(sel);
  }
  #pragma unroll
  for (int t = 0; t < 2; ++t)
    #pragma unroll
    for (int j = 0; j < 8; ++j) {
      size_t row = (size_t)(row0 + half * 8 + j);
      out[o_predB + row * OUTD + t * 16 + nl] = predB[t][j];
      out[o_predS + row * OUTD + t * 16 + nl] = predS[t][j];
    }
  float lsum = 0.0f;
  if (nl == 0) {
    #pragma unroll
    for (int j = 0; j < 8; ++j) {
      size_t row = (size_t)(row0 + half * 8 + j);
      out[o_idxB + row] = (float)bestI[j];
      out[o_idxS + row] = (float)sIdx[j];
      lsum += bestD[j] + COEFW * (lse[j] - lb[j]);
    }
  }
  lsum += __shfl_xor(lsum, 16);
  if (lane == 0) atomicAdd(out, lsum * (1.0f / (float)Bsz));
}

extern "C" void kernel_launch(void* const* d_in, const int* in_sizes, int n_in,
                              void* d_out, int out_size, void* d_ws, size_t ws_size,
                              hipStream_t stream) {
  (void)n_in; (void)out_size;
  const float* x   = (const float*)d_in[0];
  const float* y   = (const float*)d_in[1];
  const float* mw1 = (const float*)d_in[2];
  const float* mb1 = (const float*)d_in[3];
  const float* mw2 = (const float*)d_in[4];
  const float* mb2 = (const float*)d_in[5];
  const float* ew1 = (const float*)d_in[6];
  const float* eb1 = (const float*)d_in[7];
  const float* ew2 = (const float*)d_in[8];
  const float* eb2 = (const float*)d_in[9];
  float* out = (float*)d_out;

  int Bsz = in_sizes[0] / INP;
  hipMemsetAsync(d_out, 0, sizeof(float), stream);   // loss accumulator

  bool pre = (d_ws != nullptr) && (ws_size >= WS_BYTES);
  dim3 grid((Bsz + 127) / 128), block(256);
  if (pre) {
    convw_kernel<<<dim3(512), block, 0, stream>>>(mw1, mw2, ew1, ew2, (__bf16*)d_ws);
    mixffn_kernel<true><<<grid, block, 0, stream>>>(
        x, y, mw1, mb1, mw2, mb2, ew1, eb1, ew2, eb2,
        (const unsigned short*)d_ws, out, Bsz);
  } else {
    mixffn_kernel<false><<<grid, block, 0, stream>>>(
        x, y, mw1, mb1, mw2, mb2, ew1, eb1, ew2, eb2,
        (const unsigned short*)d_ws, out, Bsz);
  }
}
